// TransformerBlock_26594437496836
// MI455X (gfx1250) — compile-verified, matrix operand order checked by reading, not run on hardware
//
#include <hip/hip_runtime.h>
#include <hip/hip_bf16.h>
#include <math.h>

// ---------------------------------------------------------------------------
// Types for CDNA5 WMMA (wave32): V_WMMA_F32_16X16X32_F16
// ---------------------------------------------------------------------------
typedef __attribute__((ext_vector_type(16))) _Float16 v16h;
typedef __attribute__((ext_vector_type(8)))  _Float16 v8h;
typedef __attribute__((ext_vector_type(8)))  float    v8f;

#define HID 128
#define LTXT 32

// ---------------------------------------------------------------------------
// Fragment load from LDS (f16, row-major with row stride LDK halves).
// ISA 16-bit A 16x32 layout: lane m=lane&15, hs=lane>>4;
//   elems 0-7  -> K = 8*hs   .. 8*hs+7      (contiguous -> ds_load_b128)
//   elems 8-15 -> K = 16+8*hs .. 16+8*hs+7  (contiguous -> ds_load_b128)
// B is staged transposed ([N][K]) so the same loader serves both operands.
// ---------------------------------------------------------------------------
template<int LDK>
__device__ inline v16h ld_frag(const _Float16* __restrict__ tile, int ks, int lane) {
    int m  = lane & 15;
    int hs = lane >> 4;
    const _Float16* p = tile + m * LDK + ks + 8 * hs;
    v8h lo = *(const v8h*)(p);
    v8h hi = *(const v8h*)(p + 16);
    v16h r;
#pragma unroll
    for (int i = 0; i < 8; ++i) { r[i] = lo[i]; r[i + 8] = hi[i]; }
    return r;
}

__device__ inline float wred_sum(float v) {
#pragma unroll
    for (int o = 16; o > 0; o >>= 1) v += __shfl_xor(v, o, 32);
    return v;
}
__device__ inline float wred_max(float v) {
#pragma unroll
    for (int o = 16; o > 0; o >>= 1) v = fmaxf(v, __shfl_xor(v, o, 32));
    return v;
}

// monotone float<->int encoding for atomic max on signed ints
__device__ inline int   encf(float f) { int i = __float_as_int(f); return i >= 0 ? i : (i ^ 0x7fffffff); }
__device__ inline float decf(int i)   { return __int_as_float(i >= 0 ? i : (i ^ 0x7fffffff)); }

// ---------------------------------------------------------------------------
// Generic WMMA GEMM: C[M x Ncols] = A(f16)[M x K] * B(f32)[K x ldb] + bias
// Block: 256 thr (8 waves), tile 32 rows x 128 cols, 2 wmma-tiles per wave.
// EPI: 0 = store f32
//      1 = res + v                  (residual add)
//      2 = res + gate[row,gcol+col]*v   (gated residual, gate stride 768)
//      3 = f16(gelu_tanh(v))        (feeds next GEMM's f16 A operand)
// ---------------------------------------------------------------------------
template<int K, int EPI>
__global__ __launch_bounds__(256) void gemm16(
    const _Float16* __restrict__ A, const float* __restrict__ B, int ldb,
    const float* __restrict__ bias,
    float* __restrict__ Cf, _Float16* __restrict__ Ch, int ldc,
    const float* __restrict__ res, const float* __restrict__ gate, int gcol,
    int Mrows)
{
    constexpr int KC = 128, KP = KC + 8;
    __shared__ __align__(16) _Float16 As[32 * KP];
    __shared__ __align__(16) _Float16 Bs[128 * KP];

    const int tid  = threadIdx.x;
    const int lane = tid & 31;
    const int wave = tid >> 5;
    const int row0 = blockIdx.y * 32;
    const int col0 = blockIdx.x * 128;
    const int r    = wave >> 2;          // row tile 0..1
    const int c0   = (wave & 3) * 2;     // col tiles c0, c0+1

    v8f acc0 = {}, acc1 = {};

    for (int kb = 0; kb < K; kb += KC) {
        // stage A (f16, 32 x KC) as dword copies
        for (int i = tid; i < 32 * (KC / 2); i += 256) {
            int m  = i >> 6;
            int k2 = i & 63;
            int rr = row0 + m; if (rr >= Mrows) rr = Mrows - 1;
            ((unsigned int*)(As + m * KP))[k2] =
                *(const unsigned int*)(A + (size_t)rr * K + kb + 2 * k2);
        }
        // stage B transposed: Bs[n][k] = (f16) B[kb+k][col0+n]
        for (int i = tid; i < 128 * KC; i += 256) {
            int n = i & 127, k = i >> 7;
            Bs[n * KP + k] = (_Float16)B[(size_t)(kb + k) * ldb + col0 + n];
        }
        if (kb + KC < K)
            __builtin_prefetch(A + (size_t)row0 * K + kb + KC, 0, 0);
        __syncthreads();

#pragma unroll
        for (int ks = 0; ks < KC; ks += 32) {
            v16h a  = ld_frag<KP>(As + r * 16 * KP, ks, lane);
            v16h b0 = ld_frag<KP>(Bs + c0 * 16 * KP, ks, lane);
            acc0 = __builtin_amdgcn_wmma_f32_16x16x32_f16(false, a, false, b0,
                                                          (short)0, acc0, false, false);
            v16h b1 = ld_frag<KP>(Bs + (c0 + 1) * 16 * KP, ks, lane);
            acc1 = __builtin_amdgcn_wmma_f32_16x16x32_f16(false, a, false, b1,
                                                          (short)0, acc1, false, false);
        }
        __syncthreads();
    }

    // epilogue: C/D layout -> lane: n = lane&15, vgpr r8 : row = r8 + 8*(lane>>4)
    const int nn = lane & 15, hs = lane >> 4;
#pragma unroll
    for (int tt = 0; tt < 2; ++tt) {
        v8f acc = tt ? acc1 : acc0;
        int col = col0 + (c0 + tt) * 16 + nn;
        float bv = bias[col];
#pragma unroll
        for (int r8 = 0; r8 < 8; ++r8) {
            int row = row0 + r * 16 + r8 + 8 * hs;
            if (row >= Mrows) continue;
            float v = acc[r8] + bv;
            size_t idx = (size_t)row * ldc + col;
            if constexpr (EPI == 0) {
                Cf[idx] = v;
            } else if constexpr (EPI == 1) {
                Cf[idx] = res[idx] + v;
            } else if constexpr (EPI == 2) {
                Cf[idx] = res[idx] + gate[(size_t)row * 768 + gcol + col] * v;
            } else {
                float x3 = v * v * v;
                float g  = 0.5f * v * (1.f + tanhf(0.7978845608f * (v + 0.044715f * x3)));
                Ch[idx] = (_Float16)g;
            }
        }
    }
}

// ---------------------------------------------------------------------------
// Elementwise / LN kernels
// ---------------------------------------------------------------------------
__global__ void silu_to_f16(const float* __restrict__ t, _Float16* __restrict__ o, int total) {
    int g = blockIdx.x * 256 + threadIdx.x;
    if (g < total) {
        float v = t[g];
        o[g] = (_Float16)(v / (1.f + __expf(-v)));
    }
}

// mode 0: out = f16( modulate(LN(X), mod[:,cshift], mod[:,cscale]) )
// mode 1: x1 = X + mod[:,cgate]*agg  (stored);  out = f16( LN(x1)*gg + gb )
__global__ __launch_bounds__(256) void ln_mod(
    const float* __restrict__ X, const float* __restrict__ mod,
    int cshift, int cscale,
    const float* __restrict__ agg, int cgate,
    const float* __restrict__ gg, const float* __restrict__ gb,
    float* __restrict__ x1out, _Float16* __restrict__ out, int N, int mode)
{
    int row = blockIdx.x * 8 + (threadIdx.x >> 5);
    if (row >= N) return;
    int lane = threadIdx.x & 31;
    float4 v = *(const float4*)(X + (size_t)row * HID + lane * 4);
    if (mode == 1) {
        float4 a = *(const float4*)(agg + (size_t)row * HID + lane * 4);
        const float* gp = mod + (size_t)row * 768 + cgate + lane * 4;
        v.x += gp[0] * a.x; v.y += gp[1] * a.y; v.z += gp[2] * a.z; v.w += gp[3] * a.w;
        *(float4*)(x1out + (size_t)row * HID + lane * 4) = v;
    }
    float s = wred_sum(v.x + v.y + v.z + v.w);
    float q = wred_sum(v.x * v.x + v.y * v.y + v.z * v.z + v.w * v.w);
    float mu  = s * (1.f / 128.f);
    float var = q * (1.f / 128.f) - mu * mu;
    float rs  = rsqrtf(var + 1e-6f);
    float vals[4] = {v.x, v.y, v.z, v.w};
#pragma unroll
    for (int i = 0; i < 4; ++i) {
        int j = lane * 4 + i;
        float xn = (vals[i] - mu) * rs;
        float rv;
        if (mode == 1) rv = xn * gg[j] + gb[j];
        else {
            float sh = mod[(size_t)row * 768 + cshift + j];
            float sc = mod[(size_t)row * 768 + cscale + j];
            rv = xn * (1.f + sc) + sh;
        }
        out[(size_t)row * HID + j] = (_Float16)rv;
    }
}

// ---------------------------------------------------------------------------
// Graph attention (segmented softmax over edges) — atomic 3-pass
// ---------------------------------------------------------------------------
__global__ void init_md(int* __restrict__ maxes, float* __restrict__ dens, int total) {
    int g = blockIdx.x * 256 + threadIdx.x;
    if (g < total) { maxes[g] = (int)0x80000000; dens[g] = 0.f; }
}

__global__ void edge_logits(const int* __restrict__ ei,
                            const float* __restrict__ qb, const float* __restrict__ kb,
                            float* __restrict__ exb, int* __restrict__ maxes, int E)
{
    int g = blockIdx.x * 256 + threadIdx.x;
    if (g >= E * 4) return;
    int e = g >> 2, h = g & 3;
    int src = ei[e], dst = ei[E + e];
    const float4* qp = (const float4*)(qb + (size_t)dst * 512 + h * 128);
    const float4* kp = (const float4*)(kb + (size_t)src * 512 + h * 128);
    float s = 0.f;
#pragma unroll
    for (int i = 0; i < 32; ++i) {
        float4 a = qp[i], b = kp[i];
        s += a.x * b.x + a.y * b.y + a.z * b.z + a.w * b.w;
    }
    s *= 0.08838834764831845f;            // 1/sqrt(128)
    exb[(size_t)e * 4 + h] = s;
    atomicMax(&maxes[dst * 4 + h], encf(s));
}

__global__ void edge_exp(const int* __restrict__ ei,
                         float* __restrict__ exb, const int* __restrict__ maxes,
                         float* __restrict__ dens, int E)
{
    int g = blockIdx.x * 256 + threadIdx.x;
    if (g >= E * 4) return;
    int e = g >> 2, h = g & 3;
    int dst = ei[E + e];
    float ex = __expf(exb[(size_t)e * 4 + h] - decf(maxes[dst * 4 + h]));
    exb[(size_t)e * 4 + h] = ex;
    atomicAdd(&dens[dst * 4 + h], ex);
}

// agg[dst] += (1/H) * sum_h alpha[e,h] * v[src,h,:]   (agg pre-loaded with skip)
__global__ __launch_bounds__(256) void edge_agg(const int* __restrict__ ei,
                                                const float* __restrict__ exb,
                                                const float* __restrict__ dens,
                                                const float* __restrict__ vb,
                                                float* __restrict__ agg, int E)
{
    int e = blockIdx.x * 2 + (threadIdx.x >> 7);
    if (e >= E) return;
    int j = threadIdx.x & 127;
    int src = ei[e], dst = ei[E + e];
    float acc = 0.f;
#pragma unroll
    for (int h = 0; h < 4; ++h) {
        float w = exb[(size_t)e * 4 + h] / dens[dst * 4 + h];
        acc += w * vb[(size_t)src * 512 + h * 128 + j];
    }
    atomicAdd(&agg[(size_t)dst * HID + j], 0.25f * acc);
}

// ---------------------------------------------------------------------------
// Cross attention: 4 nodes per block. K/V weights (f16, transposed) + text
// tiles staged in LDS; per-node WMMA projections -> LDS; wave-per-head
// softmax; ctx emitted f16 for the fused o-proj GEMM.
// ---------------------------------------------------------------------------
__global__ __launch_bounds__(256) void cross_attn(
    const float* __restrict__ text,
    const float* __restrict__ kw, const float* __restrict__ kbias,
    const float* __restrict__ vw, const float* __restrict__ vbias,
    const float* __restrict__ qbuf, _Float16* __restrict__ ctx, int N)
{
    constexpr int KP = HID + 8;   // 136
    constexpr int CP = HID + 4;   // 132
    __shared__ __align__(16) _Float16 KT[128 * KP];
    __shared__ __align__(16) _Float16 VT[128 * KP];
    __shared__ __align__(16) _Float16 TX[4][LTXT * KP];
    __shared__ float KH[LTXT * CP];
    __shared__ float VH[LTXT * CP];
    __shared__ float ATT[4 * LTXT];

    const int tid = threadIdx.x, lane = tid & 31, wave = tid >> 5;
    const int n0 = blockIdx.x * 4;

    // stage transposed weights (f32 -> f16)
    for (int i = tid; i < 128 * 128; i += 256) {
        int n = i & 127, k = i >> 7;
        KT[n * KP + k] = (_Float16)kw[(size_t)k * 128 + n];
        VT[n * KP + k] = (_Float16)vw[(size_t)k * 128 + n];
    }
    // stage text tiles for 4 nodes
    for (int i = tid; i < 4 * LTXT * 128; i += 256) {
        int ni = i >> 12, rem = i & 4095, l = rem >> 7, k = rem & 127;
        int n = n0 + ni; if (n >= N) n = N - 1;
        TX[ni][l * KP + k] = (_Float16)text[((size_t)n * LTXT + l) * 128 + k];
    }
    __syncthreads();

    for (int ni = 0; ni < 4; ++ni) {
        const int n = n0 + ni;
        // 32 output tiles: proj(2) x rowtile(2) x coltile(8); 4 per wave
#pragma unroll
        for (int p4 = 0; p4 < 4; ++p4) {
            int t = wave + 8 * p4;
            int proj = t >> 4;
            int r = (t >> 3) & 1, c = t & 7;
            const _Float16* W = proj ? VT : KT;
            const float* bs   = proj ? vbias : kbias;
            float* OUT        = proj ? VH : KH;
            v8f acc = {};
#pragma unroll
            for (int ks = 0; ks < 128; ks += 32) {
                v16h a = ld_frag<KP>(&TX[ni][r * 16 * KP], ks, lane);
                v16h b = ld_frag<KP>(W + c * 16 * KP, ks, lane);
                acc = __builtin_amdgcn_wmma_f32_16x16x32_f16(false, a, false, b,
                                                             (short)0, acc, false, false);
            }
            int nn = lane & 15, hs = lane >> 4;
            int colc = c * 16 + nn;
            float bv = bs[colc];
#pragma unroll
            for (int r8 = 0; r8 < 8; ++r8)
                OUT[(r * 16 + r8 + 8 * hs) * CP + colc] = acc[r8] + bv;
        }
        __syncthreads();

        if (tid < 128) {                    // wave = head, lane = token
            int h = tid >> 5, l = tid & 31;
            const float* qp = qbuf + (size_t)min(n, N - 1) * 128 + h * 32;
            float s = 0.f;
#pragma unroll
            for (int d = 0; d < 32; ++d) s += qp[d] * KH[l * CP + h * 32 + d];
            s *= 0.17677669529663687f;      // 1/sqrt(32)
            float mx = wred_max(s);
            float ev = __expf(s - mx);
            float dn = wred_sum(ev);
            ATT[h * 32 + l] = ev / dn;
        }
        __syncthreads();
        if (tid < 128 && n < N) {           // wave = head, lane = dim
            int h = tid >> 5, d = tid & 31;
            float cx = 0.f;
#pragma unroll
            for (int l = 0; l < LTXT; ++l) cx += ATT[h * 32 + l] * VH[l * CP + h * 32 + d];
            ctx[(size_t)n * 128 + h * 32 + d] = (_Float16)cx;
        }
        __syncthreads();
    }
}

// ---------------------------------------------------------------------------
// Host launcher
// ---------------------------------------------------------------------------
extern "C" void kernel_launch(void* const* d_in, const int* in_sizes, int n_in,
                              void* d_out, int out_size, void* d_ws, size_t ws_size,
                              hipStream_t stream) {
    const float* x        = (const float*)d_in[0];
    const int*   ei       = (const int*)  d_in[1];
    const float* t        = (const float*)d_in[2];
    const float* text     = (const float*)d_in[3];
    const float* sa_q_w   = (const float*)d_in[4];
    const float* sa_q_b   = (const float*)d_in[5];
    const float* sa_k_w   = (const float*)d_in[6];
    const float* sa_k_b   = (const float*)d_in[7];
    const float* sa_v_w   = (const float*)d_in[8];
    const float* sa_v_b   = (const float*)d_in[9];
    const float* sa_sk_w  = (const float*)d_in[10];
    const float* sa_sk_b  = (const float*)d_in[11];
    const float* n3_g     = (const float*)d_in[12];
    const float* n3_b     = (const float*)d_in[13];
    const float* mha_q_w  = (const float*)d_in[14];
    const float* mha_q_b  = (const float*)d_in[15];
    const float* mha_k_w  = (const float*)d_in[16];
    const float* mha_k_b  = (const float*)d_in[17];
    const float* mha_v_w  = (const float*)d_in[18];
    const float* mha_v_b  = (const float*)d_in[19];
    const float* mha_o_w  = (const float*)d_in[20];
    const float* mha_o_b  = (const float*)d_in[21];
    const float* fc1_w    = (const float*)d_in[22];
    const float* fc1_b    = (const float*)d_in[23];
    const float* fc2_w    = (const float*)d_in[24];
    const float* fc2_b    = (const float*)d_in[25];
    const float* ada_w    = (const float*)d_in[26];
    const float* ada_b    = (const float*)d_in[27];

    const int N = in_sizes[0] / HID;
    const int E = in_sizes[1] / 2;
    float* out = (float*)d_out;

    // workspace carve-up
    size_t off = 0;
    auto alloc = [&](size_t bytes) {
        void* p = (char*)d_ws + off;
        off += (bytes + 255) & ~(size_t)255;
        return p;
    };
    float*    mod    = (float*)   alloc((size_t)N * 768 * 4);
    float*    qbuf   = (float*)   alloc((size_t)N * 512 * 4);
    float*    kbuf   = (float*)   alloc((size_t)N * 512 * 4);
    float*    vbuf   = (float*)   alloc((size_t)N * 512 * 4);
    float*    aggbuf = (float*)   alloc((size_t)N * HID * 4);
    int*      maxes  = (int*)     alloc((size_t)N * 4 * 4);
    float*    dens   = (float*)   alloc((size_t)N * 4 * 4);
    float*    exbuf  = (float*)   alloc((size_t)E * 4 * 4);
    float*    x1     = (float*)   alloc((size_t)N * HID * 4);
    float*    mhaq   = (float*)   alloc((size_t)N * HID * 4);
    float*    x2     = (float*)   alloc((size_t)N * HID * 4);
    _Float16* siluT  = (_Float16*)alloc((size_t)N * HID * 2);
    _Float16* xm     = (_Float16*)alloc((size_t)N * HID * 2);
    _Float16* qn     = (_Float16*)alloc((size_t)N * HID * 2);
    _Float16* ctx    = (_Float16*)alloc((size_t)N * HID * 2);
    _Float16* xm2    = (_Float16*)alloc((size_t)N * HID * 2);
    _Float16* hbuf   = (_Float16*)alloc((size_t)N * 512 * 2);
    (void)ws_size; (void)n_in; (void)out_size;

    const int rb = (N + 31) / 32;   // GEMM row-blocks
    const dim3 blk(256);

    // adaLN: mod = silu(t) @ ada_w + ada_b
    silu_to_f16<<<(N * HID + 255) / 256, blk, 0, stream>>>(t, siluT, N * HID);
    gemm16<128, 0><<<dim3(6, rb), blk, 0, stream>>>(siluT, ada_w, 768, ada_b,
        mod, nullptr, 768, nullptr, nullptr, 0, N);

    // xm = modulate(LN(x), shift_msa, scale_msa)
    ln_mod<<<(N + 7) / 8, blk, 0, stream>>>(x, mod, 0, 128, nullptr, 0,
        nullptr, nullptr, nullptr, xm, N, 0);

    // graph-SA projections (agg buffer seeded with root skip)
    gemm16<128, 0><<<dim3(4, rb), blk, 0, stream>>>(xm, sa_q_w, 512, sa_q_b,
        qbuf, nullptr, 512, nullptr, nullptr, 0, N);
    gemm16<128, 0><<<dim3(4, rb), blk, 0, stream>>>(xm, sa_k_w, 512, sa_k_b,
        kbuf, nullptr, 512, nullptr, nullptr, 0, N);
    gemm16<128, 0><<<dim3(4, rb), blk, 0, stream>>>(xm, sa_v_w, 512, sa_v_b,
        vbuf, nullptr, 512, nullptr, nullptr, 0, N);
    gemm16<128, 0><<<dim3(1, rb), blk, 0, stream>>>(xm, sa_sk_w, 128, sa_sk_b,
        aggbuf, nullptr, 128, nullptr, nullptr, 0, N);

    // segmented softmax over edges
    init_md<<<(N * 4 + 255) / 256, blk, 0, stream>>>(maxes, dens, N * 4);
    edge_logits<<<(E * 4 + 255) / 256, blk, 0, stream>>>(ei, qbuf, kbuf, exbuf, maxes, E);
    edge_exp   <<<(E * 4 + 255) / 256, blk, 0, stream>>>(ei, exbuf, maxes, dens, E);
    edge_agg   <<<(E + 1) / 2, blk, 0, stream>>>(ei, exbuf, dens, vbuf, aggbuf, E);

    // x1 = x + gate_msa*sa_out ; qn = LN(x1)*g + b
    ln_mod<<<(N + 7) / 8, blk, 0, stream>>>(x, mod, 0, 0, aggbuf, 256,
        n3_g, n3_b, x1, qn, N, 1);

    // mha query projection
    gemm16<128, 0><<<dim3(1, rb), blk, 0, stream>>>(qn, mha_q_w, 128, mha_q_b,
        mhaq, nullptr, 128, nullptr, nullptr, 0, N);

    // per-node cross attention over text memory
    cross_attn<<<(N + 3) / 4, blk, 0, stream>>>(text, mha_k_w, mha_k_b,
        mha_v_w, mha_v_b, mhaq, ctx, N);

    // x2 = x1 + ctx @ o_w + o_b
    gemm16<128, 1><<<dim3(1, rb), blk, 0, stream>>>(ctx, mha_o_w, 128, mha_o_b,
        x2, nullptr, 128, x1, nullptr, 0, N);

    // gated MLP
    ln_mod<<<(N + 7) / 8, blk, 0, stream>>>(x2, mod, 384, 512, nullptr, 0,
        nullptr, nullptr, nullptr, xm2, N, 0);
    gemm16<128, 3><<<dim3(4, rb), blk, 0, stream>>>(xm2, fc1_w, 512, fc1_b,
        nullptr, hbuf, 512, nullptr, nullptr, 0, N);
    gemm16<512, 2><<<dim3(1, rb), blk, 0, stream>>>(hbuf, fc2_w, 128, fc2_b,
        out, nullptr, 128, x2, mod, 640, N);
}
